// CrossModalAttention_87187836108878
// MI455X (gfx1250) — matrix operand order repaired and checked by reading, not compiled, not run on hardware
//
#include <hip/hip_runtime.h>

// CDNA5 (gfx1250, wave32) fused cross-modal attention.
// Pipeline: [proj q/k/v with v_wmma_f32_16x16x32_f16] -> [flash attention,
// online softmax, TDM(tensor_load_to_lds)-staged double-buffered K/V chunks]
// -> [output proj + residual].  f16 storage, f32 accumulation.

typedef _Float16 v16h __attribute__((ext_vector_type(16)));
typedef float v8f __attribute__((ext_vector_type(8)));
typedef unsigned int v4u __attribute__((ext_vector_type(4)));
typedef int v8i __attribute__((ext_vector_type(8)));
typedef int v4i __attribute__((ext_vector_type(4)));

union HFrag {
    v16h v;
    unsigned int u[8];
    _Float16 h[16];
};

#define WMMA_F32_F16(A, B, C) \
    __builtin_amdgcn_wmma_f32_16x16x32_f16(false, (A), false, (B), (short)0, (C), false, false)

__device__ __forceinline__ int aKoff(int r, int hi) {
    // 16-bit A-matrix 16x32 layout: per-lane K offsets, contiguous k = 16*hi + 2*r
    return 16 * hi + 2 * r;
}

#define HAVE_TDM __has_builtin(__builtin_amdgcn_tensor_load_to_lds)

#if HAVE_TDM
typedef __attribute__((address_space(3))) void lds_void_t;

__device__ __forceinline__ unsigned lds_byte_off(void* p) {
    return (unsigned)(unsigned long long)(lds_void_t*)p;
}

// 2-D tile DMA: global -> LDS, 16-bit elements (D# per ISA 8.3/8.4).
// tile rows are packed contiguously in LDS. 6-arg builtin form (clang-23 /
// therock signature): (g0 v4u, g1 v8i, g2 v4i, g3 v4i, g4 v8i, cpol).
__device__ __forceinline__ void tdm_load_2d(
    unsigned lds_addr, const void* gaddr,
    unsigned tile_d0, unsigned tile_d1,
    unsigned tensor_d0, unsigned tensor_d1,
    unsigned long long stride0 /* elements */)
{
    unsigned long long ga = (unsigned long long)gaddr;
    v4u g0;
    g0[0] = 1u;                                   // count=1 (valid user D#)
    g0[1] = lds_addr;                             // lds_addr[31:0]
    g0[2] = (unsigned)(ga & 0xffffffffu);         // global_addr[31:0]
    g0[3] = (unsigned)((ga >> 32) & 0x01ffffffu)  // global_addr[56:32]
            | (2u << 30);                         // type=2 ("image")
    v8i g1;
    g1[0] = (int)(1u << 16);                      // data_size=1 -> 2 bytes
    g1[1] = (int)((tensor_d0 & 0xffffu) << 16);   // tensor_dim0[15:0]
    g1[2] = (int)((tensor_d0 >> 16) | ((tensor_d1 & 0xffffu) << 16));
    g1[3] = (int)((tensor_d1 >> 16) | (tile_d0 << 16));   // tile_dim0
    g1[4] = (int)(tile_d1 & 0xffffu);             // tile_dim1 (tile_dim2=0)
    g1[5] = (int)(stride0 & 0xffffffffu);         // tensor_dim0_stride[31:0]
    g1[6] = (int)((stride0 >> 32) & 0xffffu);     // stride[47:32]; dim1_stride=0
    g1[7] = 0;
    v4i z4 = {0, 0, 0, 0};                        // groups 2/3: unused (2-D)
    v8i z8 = {0, 0, 0, 0, 0, 0, 0, 0};            // extra group: unused
    __builtin_amdgcn_tensor_load_to_lds(g0, g1, z4, z4, z8, 0);
}
#endif

// ---------------------------------------------------------------------------
// Kernel 1: channel projection (1x1 conv == GEMM).
//   out[o,n] = W[o,:] . X[:,n] + bias[o]        (per batch)
// transposeOut=1 -> store as [N, O] f16;  0 -> [O, N] f16
// ---------------------------------------------------------------------------
__global__ __launch_bounds__(256) void proj_qkv(
    const float* __restrict__ X,     // [4, C, 4096]
    const float* __restrict__ W,     // [O, C]
    const float* __restrict__ bias,  // [O]
    _Float16* __restrict__ out,
    int C, int Otiles, float outScale, int transposeOut)
{
    const int lane  = threadIdx.x & 31;
    const int wave  = threadIdx.x >> 5;
    const int tile  = blockIdx.x * 8 + wave;
    const int total = 4 * Otiles * 256;
    if (tile >= total) return;

    const int ntile = tile & 255;
    int rest = tile >> 8;
    const int otile = rest % Otiles;
    const int b     = rest / Otiles;
    const int O  = Otiles * 16;
    const int n0 = ntile * 16;
    const int o0 = otile * 16;
    const int li = lane & 15;
    const int hi = lane >> 4;

    const float* Xb = X + (size_t)b * C * 4096;
    v8f acc = {};
    const int nChunks = C >> 5;
    for (int cc = 0; cc < nChunks; ++cc) {
        HFrag a, bf;
        const float* wrow = W + (size_t)(o0 + li) * C + cc * 32;
#pragma unroll
        for (int r = 0; r < 8; ++r) {
            int k = aKoff(r, hi);
            a.h[2 * r]     = (_Float16)wrow[k];
            a.h[2 * r + 1] = (_Float16)wrow[k + 1];
        }
        const float* xcol = Xb + (size_t)(cc * 32) * 4096 + n0 + li;
#pragma unroll
        for (int vv = 0; vv < 8; ++vv) {
            int c = 2 * vv + 16 * hi;
            bf.h[2 * vv]     = (_Float16)xcol[(size_t)c * 4096];
            bf.h[2 * vv + 1] = (_Float16)xcol[(size_t)(c + 1) * 4096];
        }
        acc = WMMA_F32_F16(a.v, bf.v, acc);
    }

    if (transposeOut) {
        _Float16* orow = out + (size_t)b * 4096 * O + (size_t)(n0 + li) * O + o0 + 8 * hi;
#pragma unroll
        for (int r = 0; r < 8; r += 2) {
            union { unsigned int u; _Float16 h[2]; } p;
            p.h[0] = (_Float16)((acc[r]     + bias[o0 + r     + 8 * hi]) * outScale);
            p.h[1] = (_Float16)((acc[r + 1] + bias[o0 + r + 1 + 8 * hi]) * outScale);
            *(unsigned int*)&orow[r] = p.u;
        }
    } else {
#pragma unroll
        for (int r = 0; r < 8; ++r) {
            int o = o0 + r + 8 * hi;
            out[(size_t)b * O * 4096 + (size_t)o * 4096 + n0 + li] =
                (_Float16)((acc[r] + bias[o]) * outScale);
        }
    }
}

// ---------------------------------------------------------------------------
// Kernel 2: flash attention. 4 waves/block, 16 queries/wave, 32-key chunks
// double-buffered in LDS via the Tensor Data Mover (TENSORcnt-tracked),
// overlapping DMA of chunk c+1 with the 36 WMMAs of chunk c.
// Sᵀ = Kᵀ·Q so softmax is lane-local + one shfl_xor(16); O = V·Pᵀ over 20
// Cu-tiles with online rescale. qT pre-scaled by D^-0.5. Output transposed.
// ---------------------------------------------------------------------------
__global__ __launch_bounds__(128) void flash_attn(
    const _Float16* __restrict__ qT,   // [4, 4096, 256]  (pre-scaled)
    const _Float16* __restrict__ kT,   // [4, 4096, 256]
    const _Float16* __restrict__ vv,   // [4, 320, 4096]
    _Float16* __restrict__ attnT)      // [4, 4096, 320]
{
    __shared__ _Float16 lds_k[2][32 * 256];   // 2 x 16 KB
    __shared__ _Float16 lds_v[2][320 * 32];   // 2 x 20 KB
    __shared__ _Float16 lds_p[4][16 * 32];    //     4 KB per-wave P transpose

    const int tid  = threadIdx.x;
    const int lane = tid & 31;
    const int wave = tid >> 5;
    const int b    = blockIdx.x >> 6;                       // 64 blocks/batch
    const int i0   = ((blockIdx.x & 63) * 4 + wave) * 16;   // query tile base
    const int li   = lane & 15;
    const int hi   = lane >> 4;

    const _Float16* qTb = qT + (size_t)b * 4096 * 256;
    const _Float16* kTb = kT + (size_t)b * 4096 * 256;
    const _Float16* vb  = vv + (size_t)b * 320 * 4096;

    // persistent Q fragments (B operand: [ch32 x query16]) for 8 ch-chunks
    unsigned int qf[8][8];
    {
        const unsigned int* qrow = (const unsigned int*)(qTb + (size_t)(i0 + li) * 256);
#pragma unroll
        for (int cc = 0; cc < 8; ++cc)
#pragma unroll
            for (int r = 0; r < 8; ++r)
                qf[cc][r] = qrow[(cc * 32 + 2 * r + 16 * hi) >> 1];
    }

    v8f acc[20];
#pragma unroll
    for (int t = 0; t < 20; ++t) acc[t] = (v8f){};
    float mrow = -1e30f, lrow = 0.0f;

#if HAVE_TDM
    if (wave == 0) {   // prime chunk 0 (TDM: EXEC-ignored, per-wave issue)
        tdm_load_2d(lds_byte_off(&lds_k[0][0]), kTb, 256, 32, 256, 4096, 256);
        tdm_load_2d(lds_byte_off(&lds_v[0][0]), vb,  32, 320, 4096, 320, 4096);
    }
#endif

    for (int c = 0; c < 128; ++c) {
        const int p  = c & 1;
        const int j0 = c * 32;

#if HAVE_TDM
        if (wave == 0) __builtin_amdgcn_s_wait_tensorcnt(0);
        __syncthreads();                 // buffer p is now filled & visible
        if (c + 1 < 128 && wave == 0) {  // overlap DMA of chunk c+1
            const int jn = j0 + 32;
            tdm_load_2d(lds_byte_off(&lds_k[p ^ 1][0]),
                        kTb + (size_t)jn * 256, 256, 32, 256, 4096, 256);
            tdm_load_2d(lds_byte_off(&lds_v[p ^ 1][0]),
                        vb + jn, 32, 320, 4096, 320, 4096);
        }
#else
        __syncthreads();                 // all waves done with buffer p
        {                                // stage K chunk: contiguous 16 KB
            const uint4* src = (const uint4*)(kTb + (size_t)j0 * 256);
            uint4* dst = (uint4*)&lds_k[p][0];
#pragma unroll
            for (int k = 0; k < 8; ++k) dst[tid + k * 128] = src[tid + k * 128];
        }
        {                                // stage V chunk: 320 rows x 64 B
            uint4* dst = (uint4*)&lds_v[p][0];
            for (int idx = tid; idx < 1280; idx += 128) {
                int cch = idx >> 2, q = idx & 3;
                dst[idx] = *(const uint4*)(vb + (size_t)cch * 4096 + j0 + q * 8);
            }
        }
        __syncthreads();
#endif

        // S^T tiles: [key16 x query16] x2, K-loop over 256 channels
        v8f st[2]; st[0] = (v8f){}; st[1] = (v8f){};
#pragma unroll
        for (int sub = 0; sub < 2; ++sub) {
            const unsigned int* krow =
                (const unsigned int*)(&lds_k[p][0] + (size_t)(sub * 16 + li) * 256);
#pragma unroll
            for (int cc = 0; cc < 8; ++cc) {
                HFrag a, qb;
#pragma unroll
                for (int r = 0; r < 8; ++r)
                    a.u[r] = krow[(cc * 32 + aKoff(r, hi)) >> 1];
#pragma unroll
                for (int r = 0; r < 8; ++r) qb.u[r] = qf[cc][r];
                st[sub] = WMMA_F32_F16(a.v, qb.v, st[sub]);
            }
        }

        // online softmax: lane holds 16 key-scores of query li; partner
        // (lane^16) holds the other 16
        float mc = -1e30f;
#pragma unroll
        for (int r = 0; r < 8; ++r) {
            mc = fmaxf(mc, st[0][r]);
            mc = fmaxf(mc, st[1][r]);
        }
        mc = fmaxf(mc, __shfl_xor(mc, 16, 32));
        const float mnew  = fmaxf(mrow, mc);
        const float alpha = __expf(mrow - mnew);

        float rs = 0.0f;
        _Float16 ph[16];
#pragma unroll
        for (int sub = 0; sub < 2; ++sub)
#pragma unroll
            for (int r = 0; r < 8; ++r) {
                float pe = __expf(st[sub][r] - mnew);
                rs += pe;
                ph[sub * 8 + r] = (_Float16)pe;
            }
        rs += __shfl_xor(rs, 16, 32);
        lrow = lrow * alpha + rs;
        mrow = mnew;

#pragma unroll
        for (int t = 0; t < 20; ++t)
#pragma unroll
            for (int e = 0; e < 8; ++e) acc[t][e] *= alpha;

        // transpose P through per-wave LDS into B-operand layout
        _Float16* prow = lds_p[wave] + li * 32;
#pragma unroll
        for (int sub = 0; sub < 2; ++sub)
#pragma unroll
            for (int r = 0; r < 8; ++r)
                prow[sub * 16 + r + 8 * hi] = ph[sub * 8 + r];
        asm volatile("s_wait_dscnt 0" ::: "memory");  // same-wave cross-lane RAW

        HFrag pf;
#pragma unroll
        for (int r = 0; r < 8; ++r)
            pf.u[r] = *(const unsigned int*)(lds_p[wave] + li * 32 + 2 * r + 16 * hi);

        // O += V x P^T over 20 Cu-tiles
#pragma unroll
        for (int t = 0; t < 20; ++t) {
            HFrag a;
            const unsigned int* vrow =
                (const unsigned int*)(&lds_v[p][0] + (size_t)(t * 16 + li) * 32);
#pragma unroll
            for (int r = 0; r < 8; ++r) a.u[r] = vrow[aKoff(r, hi) >> 1];
            acc[t] = WMMA_F32_F16(a.v, pf.v, acc[t]);
        }
    }

    // normalize + store transposed: attnT[n, c]
    const float rl = 1.0f / lrow;
    _Float16* orow = attnT + (size_t)b * 4096 * 320 + (size_t)(i0 + li) * 320;
#pragma unroll
    for (int t = 0; t < 20; ++t)
#pragma unroll
        for (int r = 0; r < 8; r += 2) {
            union { unsigned int u; _Float16 h[2]; } pk;
            pk.h[0] = (_Float16)(acc[t][r]     * rl);
            pk.h[1] = (_Float16)(acc[t][r + 1] * rl);
            *(unsigned int*)&orow[t * 16 + 8 * hi + r] = pk.u;
        }
}

// ---------------------------------------------------------------------------
// Kernel 3: out = Wo x attn_out + bo + residual   (f32 output)
// ---------------------------------------------------------------------------
__global__ __launch_bounds__(256) void out_proj(
    const _Float16* __restrict__ attnT,  // [4, 4096, 320]
    const float* __restrict__ Wo,        // [320, 320]
    const float* __restrict__ bo,        // [320]
    const float* __restrict__ unet,      // [4, 320, 4096]
    float* __restrict__ out)             // [4, 320, 4096]
{
    const int lane = threadIdx.x & 31;
    const int wave = threadIdx.x >> 5;
    const int tile = blockIdx.x * 8 + wave;  // 4*20*256 = 20480 tiles
    if (tile >= 4 * 20 * 256) return;

    const int ntile = tile & 255;
    int rest = tile >> 8;
    const int otile = rest % 20;
    const int b     = rest / 20;
    const int n0 = ntile * 16, o0 = otile * 16;
    const int li = lane & 15, hi = lane >> 4;

    const _Float16* aT = attnT + (size_t)b * 4096 * 320;
    const unsigned int* arow = (const unsigned int*)(aT + (size_t)(n0 + li) * 320);

    v8f acc = {};
#pragma unroll
    for (int cc = 0; cc < 10; ++cc) {
        HFrag a, bf;
        const float* wrow = Wo + (size_t)(o0 + li) * 320 + cc * 32;
#pragma unroll
        for (int r = 0; r < 8; ++r) {
            int k = aKoff(r, hi);
            a.h[2 * r]     = (_Float16)wrow[k];
            a.h[2 * r + 1] = (_Float16)wrow[k + 1];
        }
#pragma unroll
        for (int r = 0; r < 8; ++r)
            bf.u[r] = arow[(cc * 32 + 2 * r + 16 * hi) >> 1];
        acc = WMMA_F32_F16(a.v, bf.v, acc);
    }

    const size_t base = (size_t)b * 320 * 4096;
#pragma unroll
    for (int r = 0; r < 8; ++r) {
        int o = o0 + r + 8 * hi;
        size_t idx = base + (size_t)o * 4096 + n0 + li;
        out[idx] = acc[r] + bo[o] + unet[idx];
    }
}

// ---------------------------------------------------------------------------
extern "C" void kernel_launch(void* const* d_in, const int* in_sizes, int n_in,
                              void* d_out, int out_size, void* d_ws, size_t ws_size,
                              hipStream_t stream) {
    const float* unet  = (const float*)d_in[0];
    const float* janus = (const float*)d_in[1];
    const float* Wq    = (const float*)d_in[2];
    const float* bq    = (const float*)d_in[3];
    const float* Wk    = (const float*)d_in[4];
    const float* bk    = (const float*)d_in[5];
    const float* Wv    = (const float*)d_in[6];
    const float* bv    = (const float*)d_in[7];
    const float* Wo    = (const float*)d_in[8];
    const float* bo    = (const float*)d_in[9];
    float* out = (float*)d_out;

    // workspace layout (f16): qT 8MB | kT 8MB | v 10MB | attnT 10MB = 36 MB
    _Float16* qT = (_Float16*)d_ws;
    _Float16* kT = qT + (size_t)4 * 4096 * 256;
    _Float16* vB = kT + (size_t)4 * 4096 * 256;
    _Float16* aT = vB + (size_t)4 * 320 * 4096;

    const float scale = 0.0625f;  // 256^-0.5, folded into q

    proj_qkv<<<2048, 256, 0, stream>>>(unet,  Wq, bq, qT, 320,  16, scale, 1);
    proj_qkv<<<2048, 256, 0, stream>>>(janus, Wk, bk, kT, 1024, 16, 1.0f,  1);
    proj_qkv<<<2560, 256, 0, stream>>>(janus, Wv, bv, vB, 1024, 20, 1.0f,  0);
    flash_attn<<<256, 128, 0, stream>>>(qT, kT, vB, aT);
    out_proj<<<2560, 256, 0, stream>>>(aT, Wo, bo, unet, out);
}
